// REDSwinSA_43061342109714
// MI455X (gfx1250) — matrix operand order repaired and checked by reading, not compiled, not run on hardware
//
#include <hip/hip_runtime.h>
#include <hip/hip_bf16.h>

typedef __attribute__((ext_vector_type(16))) _Float16 v16h;
typedef __attribute__((ext_vector_type(8)))  _Float16 v8h;
typedef __attribute__((ext_vector_type(8)))  float    v8f;

#define LDSTR       528                       // halves per row (512 + 16 pad)
#define REG_BYTES   (64 * LDSTR * 2)          // 67584 B per 64x512 f16 region
#define DE_OFF      (4 * REG_BYTES)           // 270336
#define IDX_OFF     (DE_OFF + 2032 * 4)       // 278464
#define SMEM_BYTES  (IDX_OFF + 64 * 4)        // 278720  (< 320 KB/WGP)

#define WMMA(a, b, c) __builtin_amdgcn_wmma_f32_16x16x32_f16(false, (a), false, (b), (short)0, (c), false, false)

// A fragment (16x32 f16, M x K). Layout: lanes 0-15 -> M, halves 0..15 = K 0..15;
// lanes 16-31: K 16..31.
__device__ __forceinline__ v16h load_a_frag(const _Float16* p0, int strideH, int lane) {
  const int m    = lane & 15;
  const int koff = (lane & 16) ? 16 : 0;
  const _Float16* p = p0 + m * strideH + koff;
  v8h lo = *(const v8h*)(p);
  v8h hi = *(const v8h*)(p + 8);
  v16h r;
#pragma unroll
  for (int i = 0; i < 8; ++i) { r[i] = lo[i]; r[i + 8] = hi[i]; }
  return r;
}

// B fragment (32x16 f16, K x N), source laid out [N][K] contiguous along K.
// lanes 0-15: N=lane, K=0..15 in halves; lanes 16-31: K=16..31.
__device__ __forceinline__ v16h load_b_frag(const _Float16* p0, int strideH, int lane) {
  const int n    = lane & 15;
  const int koff = (lane & 16) ? 16 : 0;
  return *(const v16h*)(p0 + n * strideH + koff);
}

// ---- weight conversion f32 -> f16 into workspace: [wq|wk|wv|wo], each 512x512 ----
__global__ void prep_weights_kernel(const float* __restrict__ qw, const float* __restrict__ kw,
                                    const float* __restrict__ vw, const float* __restrict__ ow,
                                    _Float16* __restrict__ w16) {
  const int i = blockIdx.x * 256 + threadIdx.x;     // 0 .. 4*262144-1
  const int N = 512 * 512;
  const float* src = (i < N) ? qw : (i < 2 * N) ? kw : (i < 3 * N) ? vw : ow;
  w16[i] = (_Float16)src[i & (N - 1)];
}

// ---- fused per-window shifted-window attention ----
__global__ __launch_bounds__(256)
void swin_window_kernel(const float* __restrict__ x,
                        const int*   __restrict__ indices,
                        const float* __restrict__ norm_w, const float* __restrict__ norm_b,
                        const float* __restrict__ q_b,    const float* __restrict__ k_b,
                        const float* __restrict__ v_b,    const float* __restrict__ o_b,
                        const float* __restrict__ depth_embedding,
                        const _Float16* __restrict__ w16,
                        float* __restrict__ out) {
  const int tid    = threadIdx.x;
  const int lane   = tid & 31;
  const int waveId = tid >> 5;
  const int w  = blockIdx.x;          // window id, 0..2047
  const int b  = w >> 8;
  const int wy = (w >> 4) & 15;
  const int wx = w & 15;

  extern __shared__ char smem_raw[];
  _Float16* XN  = (_Float16*)(smem_raw);                   // later reused as O (64x512)
  _Float16* Qs  = (_Float16*)(smem_raw + REG_BYTES);       // later reused as per-wave P scratch
  _Float16* Ks  = (_Float16*)(smem_raw + 2 * REG_BYTES);
  _Float16* Vt  = (_Float16*)(smem_raw + 3 * REG_BYTES);   // V TRANSPOSED: [d][t], stride 64
  float*    deL  = (float*)(smem_raw + DE_OFF);
  int*      idxL = (int*)(smem_raw + IDX_OFF);

  // stage depth-embedding table (127x16 f32) + shifted window indices
  for (int i = tid; i < 127 * 16; i += 256) deL[i] = depth_embedding[i];
  if (tid < 64) {
    const int ty = tid >> 3, tx = tid & 7;
    const int gy = (wy * 8 + ty + 4) & 127;
    const int gx = (wx * 8 + tx + 4) & 127;
    idxL[tid] = indices[((size_t)b * 128 + gy) * 128 + gx];
  }

  // ---- Phase 1: load shifted window + LayerNorm -> XN (f16) ----
  // each lane owns a contiguous 16-element chunk: d = lane*16 .. lane*16+15
  float nwv[16], nbv[16];
  {
    const float4* nw4 = (const float4*)norm_w;
    const float4* nb4 = (const float4*)norm_b;
#pragma unroll
    for (int c = 0; c < 4; ++c) {
      const float4 a = nw4[lane * 4 + c];
      nwv[c * 4 + 0] = a.x; nwv[c * 4 + 1] = a.y; nwv[c * 4 + 2] = a.z; nwv[c * 4 + 3] = a.w;
      const float4 bb = nb4[lane * 4 + c];
      nbv[c * 4 + 0] = bb.x; nbv[c * 4 + 1] = bb.y; nbv[c * 4 + 2] = bb.z; nbv[c * 4 + 3] = bb.w;
    }
  }

  for (int i = 0; i < 8; ++i) {
    const int t  = waveId * 8 + i;
    const int ty = t >> 3, tx = t & 7;
    const int gy = (wy * 8 + ty + 4) & 127;
    const int gx = (wx * 8 + tx + 4) & 127;
    const float4* src4 = (const float4*)(x + (((size_t)b * 128 + gy) * 128 + gx) * 512);
    float v[16];
    float s = 0.f;
#pragma unroll
    for (int c = 0; c < 4; ++c) {
      const float4 f = src4[lane * 4 + c];
      v[c * 4 + 0] = f.x; v[c * 4 + 1] = f.y; v[c * 4 + 2] = f.z; v[c * 4 + 3] = f.w;
      s += f.x + f.y + f.z + f.w;
    }
#pragma unroll
    for (int m = 16; m >= 1; m >>= 1) s += __shfl_xor(s, m, 32);
    const float mean = s * (1.0f / 512.0f);
    float vs = 0.f;
#pragma unroll
    for (int j = 0; j < 16; ++j) { const float d = v[j] - mean; vs += d * d; }
#pragma unroll
    for (int m = 16; m >= 1; m >>= 1) vs += __shfl_xor(vs, m, 32);
    const float rstd = rsqrtf(vs * (1.0f / 512.0f) + 1e-5f);
    _Float16* dst = XN + t * LDSTR + lane * 16;
#pragma unroll
    for (int c = 0; c < 2; ++c) {
      v8h pk;
#pragma unroll
      for (int j = 0; j < 8; ++j) {
        const int e = c * 8 + j;
        pk[j] = (_Float16)((v[e] - mean) * rstd * nwv[e] + nbv[e]);
      }
      *(v8h*)(dst + c * 8) = pk;
    }
  }
  __syncthreads();

  // ---- Phase 2: Q,K,V projections ----
  // Wave owns M-tile mt = waveId>>1 and N half-range; A fragments loaded ONCE,
  // reused across all 3 weight matrices and 16 column tiles (48 jobs).
  const float scale = 0.17677669529663687f;  // 1/sqrt(32)
  const int   mtw   = waveId >> 1;           // 0..3
  const int   ntb   = (waveId & 1) * 16;     // 0 or 16
  const int   n     = lane & 15;
  const int   rhalf = (lane & 16) ? 8 : 0;
  const int   row0  = mtw * 16 + rhalf;

  {
    v16h afr[16];
#pragma unroll
    for (int ks = 0; ks < 16; ++ks)
      afr[ks] = load_a_frag(XN + (mtw * 16) * LDSTR + ks * 32, LDSTR, lane);

    for (int mat = 0; mat < 3; ++mat) {
      const _Float16* W   = w16 + (size_t)mat * 262144;
      const float*   bias = (mat == 0) ? q_b : ((mat == 1) ? k_b : v_b);
      for (int ntl = 0; ntl < 16; ++ntl) {
        const int nt = ntb + ntl;
        __builtin_prefetch(W + (size_t)(nt * 16) * 512, 0, 0);
        v8f acc = {};
#pragma unroll
        for (int ks = 0; ks < 16; ++ks) {
          v16h bm = load_b_frag(W + (size_t)(nt * 16) * 512 + ks * 32, 512, lane);
          acc = WMMA(afr[ks], bm, acc);
        }
        const int   col  = nt * 16 + n;
        const float bcol = bias[col];
        if (mat == 2) {
          // V transposed: Vt[col][t], rows g are consecutive tokens -> one packed 16B store
          v8h pk;
#pragma unroll
          for (int g = 0; g < 8; ++g) pk[g] = (_Float16)(acc[g] + bcol);
          *(v8h*)(Vt + col * 64 + row0) = pk;
        } else {
          _Float16* dst = (mat == 0) ? Qs : Ks;
#pragma unroll
          for (int g = 0; g < 8; ++g) {
            float vv = acc[g] + bcol;
            if (mat == 0) vv *= scale;
            dst[(row0 + g) * LDSTR + col] = (_Float16)vv;
          }
        }
      }
    }
  }
  __syncthreads();

  // ---- Phase 3: attention, each wave handles heads {waveId, waveId+8} ----
  v16h aq[2][4];
#pragma unroll
  for (int hh = 0; hh < 2; ++hh) {
    const int h = waveId + hh * 8;
#pragma unroll
    for (int mt = 0; mt < 4; ++mt)
      aq[hh][mt] = load_a_frag(Qs + (mt * 16) * LDSTR + h * 32, LDSTR, lane);
  }
  __syncthreads();   // Qs region becomes per-wave P scratch; XN region becomes O

  _Float16* Pw = Qs + waveId * 4096;                      // 64x64 f16, stride 64
  float* attnOut = out + (size_t)8 * 128 * 128 * 512;     // attn output region
  const v8f vzero = {};

  for (int hh = 0; hh < 2; ++hh) {
    const int h = waveId + hh * 8;

    // S = Q_h (64x32) * K_h^T : K-dim = hd = 32 -> single WMMA per tile
    v8f sc[4][4];
#pragma unroll
    for (int nt = 0; nt < 4; ++nt) {
      v16h bk = load_b_frag(Ks + (nt * 16) * LDSTR + h * 32, LDSTR, lane);
#pragma unroll
      for (int mt = 0; mt < 4; ++mt) sc[mt][nt] = WMMA(aq[hh][mt], bk, vzero);
    }

    int ik[4];
#pragma unroll
    for (int nt = 0; nt < 4; ++nt) ik[nt] = idxL[nt * 16 + n];

    // depth-bias + row softmax (rows live in a 16-lane half-group)
#pragma unroll
    for (int mt = 0; mt < 4; ++mt) {
#pragma unroll
      for (int g = 0; g < 8; ++g) {
        const int q  = mt * 16 + rhalf + g;
        const int iq = idxL[q];
        float vals[4];
#pragma unroll
        for (int nt = 0; nt < 4; ++nt)
          vals[nt] = sc[mt][nt][g] + deL[(iq - ik[nt] + 63) * 16 + h];
        float mx = fmaxf(fmaxf(vals[0], vals[1]), fmaxf(vals[2], vals[3]));
#pragma unroll
        for (int m = 8; m >= 1; m >>= 1) mx = fmaxf(mx, __shfl_xor(mx, m, 16));
        float ssum = 0.f;
#pragma unroll
        for (int nt = 0; nt < 4; ++nt) { vals[nt] = __expf(vals[nt] - mx); ssum += vals[nt]; }
#pragma unroll
        for (int m = 8; m >= 1; m >>= 1) ssum += __shfl_xor(ssum, m, 16);
        const float inv = 1.0f / ssum;
        float* arow = attnOut + (((size_t)w * 16 + h) * 64 + q) * 64;
#pragma unroll
        for (int nt = 0; nt < 4; ++nt) {
          const float p = vals[nt] * inv;
          arow[nt * 16 + n]        = p;              // attn output (f32)
          Pw[q * 64 + nt * 16 + n] = (_Float16)p;    // probs for AV (f16, LDS)
        }
      }
    }

    asm volatile("s_wait_dscnt 0" ::: "memory");  // P stores visible before frag loads

    // O_h = P (64x64) * V_h (64x32) : K split into 2 steps of 32.
    // Vt is [d][t] so B fragments are contiguous loads.
    v16h bv[2][2];
#pragma unroll
    for (int nt2 = 0; nt2 < 2; ++nt2)
#pragma unroll
      for (int ks = 0; ks < 2; ++ks)
        bv[nt2][ks] = load_b_frag(Vt + (h * 32 + nt2 * 16) * 64 + ks * 32, 64, lane);

#pragma unroll
    for (int mt = 0; mt < 4; ++mt) {
      v16h ap0 = load_a_frag(Pw + (mt * 16) * 64,      64, lane);
      v16h ap1 = load_a_frag(Pw + (mt * 16) * 64 + 32, 64, lane);
#pragma unroll
      for (int nt2 = 0; nt2 < 2; ++nt2) {
        v8f o = WMMA(ap0, bv[nt2][0], vzero);
        o     = WMMA(ap1, bv[nt2][1], o);
#pragma unroll
        for (int g = 0; g < 8; ++g)
          XN[(mt * 16 + rhalf + g) * LDSTR + h * 32 + nt2 * 16 + n] = (_Float16)o[g];
      }
    }
  }
  __syncthreads();

  // ---- Phase 4: output projection + bias + residual, reverse shift ----
  // Same owner decomposition: A fragments (O tiles) loaded once per wave.
  const _Float16* Wo = w16 + (size_t)3 * 262144;
  {
    v16h afr[16];
#pragma unroll
    for (int ks = 0; ks < 16; ++ks)
      afr[ks] = load_a_frag(XN + (mtw * 16) * LDSTR + ks * 32, LDSTR, lane);

    for (int ntl = 0; ntl < 16; ++ntl) {
      const int nt = ntb + ntl;
      __builtin_prefetch(Wo + (size_t)(nt * 16) * 512, 0, 0);
      v8f acc = {};
#pragma unroll
      for (int ks = 0; ks < 16; ++ks) {
        v16h bm = load_b_frag(Wo + (size_t)(nt * 16) * 512 + ks * 32, 512, lane);
        acc = WMMA(afr[ks], bm, acc);
      }
      const int   col = nt * 16 + n;
      const float bo  = o_b[col];
#pragma unroll
      for (int g = 0; g < 8; ++g) {
        const int t  = mtw * 16 + rhalf + g;
        const int ty = t >> 3, tx = t & 7;
        const int gy = (wy * 8 + ty + 4) & 127;
        const int gx = (wx * 8 + tx + 4) & 127;
        const size_t gaddr = (((size_t)b * 128 + gy) * 128 + gx) * 512 + col;
        out[gaddr] = acc[g] + bo + x[gaddr];   // + residual at original coords
      }
    }
  }
}

extern "C" void kernel_launch(void* const* d_in, const int* in_sizes, int n_in,
                              void* d_out, int out_size, void* d_ws, size_t ws_size,
                              hipStream_t stream) {
  (void)in_sizes; (void)n_in; (void)out_size; (void)ws_size;
  const float* x       = (const float*)d_in[0];
  const int*   indices = (const int*)  d_in[1];
  const float* norm_w  = (const float*)d_in[2];
  const float* norm_b  = (const float*)d_in[3];
  const float* q_w     = (const float*)d_in[4];
  const float* q_b     = (const float*)d_in[5];
  const float* k_w     = (const float*)d_in[6];
  const float* k_b     = (const float*)d_in[7];
  const float* v_w     = (const float*)d_in[8];
  const float* v_b     = (const float*)d_in[9];
  const float* o_w     = (const float*)d_in[10];
  const float* o_b     = (const float*)d_in[11];
  const float* de      = (const float*)d_in[12];
  float*       out     = (float*)d_out;
  _Float16*    w16     = (_Float16*)d_ws;   // 4 * 512*512 f16 = 2 MB

  prep_weights_kernel<<<4096, 256, 0, stream>>>(q_w, k_w, v_w, o_w, w16);

  (void)hipFuncSetAttribute((const void*)swin_window_kernel,
                            hipFuncAttributeMaxDynamicSharedMemorySize, SMEM_BYTES);
  swin_window_kernel<<<2048, 256, SMEM_BYTES, stream>>>(
      x, indices, norm_w, norm_b, q_b, k_b, v_b, o_b, de, w16, out);
}
